// SeparatedNNxSpace_47914655154670
// MI455X (gfx1250) — matrix operand order repaired and checked by reading, not compiled, not run on hardware
//
#include <hip/hip_runtime.h>

// SeparatedNNxSpace: B=131072 samples, 8 independent MLPs (1->128->128->128),
// product over the 8 subnets, sum over rank 32 -> (B, 4).
// Compute-bound (~69 GFLOP); v_wmma_f32_16x16x32_f16 with f32 accumulate,
// hardware v_tanh_f32, double-buffered weight staging overlapped with WMMA.

#define H 128
#define LDSK 136          // padded row stride (halfs): 272B rows, conflict-friendly
#define WAVES 8

typedef __attribute__((ext_vector_type(16))) _Float16 v16h;
typedef __attribute__((ext_vector_type(8)))  _Float16 v8h;
typedef __attribute__((ext_vector_type(8)))  float    v8f;

union FragH { v16h v; v8h h[2]; };

__device__ __forceinline__ float fast_tanh(float x) {
#if __has_builtin(__builtin_amdgcn_tanhf)
  return __builtin_amdgcn_tanhf(x);   // gfx1250 V_TANH_F32, single trans op
#else
  return tanhf(x);
#endif
}

__global__ __launch_bounds__(256)
void sepnn_kernel(const float* __restrict__ features,
                  const float* __restrict__ W1, const float* __restrict__ b1,
                  const float* __restrict__ W2, const float* __restrict__ b2,
                  const float* __restrict__ W3, const float* __restrict__ b3,
                  float* __restrict__ out)
{
  __shared__ _Float16 sWA[H * LDSK];           // W2[i]^T  [n][k]
  __shared__ _Float16 sWB[H * LDSK];           // W3[i]^T  [n][k]
  __shared__ _Float16 sH1[WAVES][16 * LDSK];   // per-wave activations [m][k]
  __shared__ _Float16 sH2[WAVES][16 * LDSK];
  __shared__ float sW1[H], sB1[H], sB2[H], sB3[H];

  const int tid  = threadIdx.x;
  const int wave = tid >> 5;
  const int lane = tid & 31;
  const int half = lane >> 4;      // K-half / M-half selector
  const int lidx = lane & 15;
  const int rowBase = blockIdx.x * (WAVES * 16) + wave * 16;

  _Float16* h1 = sH1[wave];
  _Float16* h2 = sH2[wave];

  // stage a 128x128 f32 row-major matrix into LDS transposed f16 [n][k]
  auto stageW = [&](const float* __restrict__ g, _Float16* buf) {
    for (int p = tid; p < (H * H) / 8; p += 256) {
      int n  = p & 127;
      int kb = (p >> 7) * 8;
      v8h pack;
      #pragma unroll
      for (int u = 0; u < 8; ++u) pack[u] = (_Float16)g[(kb + u) * H + n];
      *(v8h*)(&buf[n * LDSK + kb]) = pack;
    }
  };

  float prod[8][8];                // [n-tile][c-row]: running product over the 8 subnets
  #pragma unroll
  for (int t = 0; t < 8; ++t)
    #pragma unroll
    for (int r = 0; r < 8; ++r) prod[t][r] = 1.0f;

  // prologue: stage W2[0] + layer-1/2 params
  stageW(W2, sWA);
  if (tid < H) {
    sW1[tid] = W1[tid];
    sB1[tid] = b1[tid];
    sB2[tid] = b2[tid];
  }
  __syncthreads();

  for (int i = 0; i < 8; ++i) {
    // ---- layer 1: h1[m][k] = tanh(x[m,i] * W1[k] + b1[k]) (rank-1, VALU) ----
    // wave-local LDS round-trip; no block barrier needed before layer 2.
    {
      int m  = lane >> 1;
      int kb = (lane & 1) * 64;
      float xv = features[(size_t)(rowBase + m) * 8 + i];
      #pragma unroll
      for (int kk = 0; kk < 64; kk += 8) {
        v8h pack;
        #pragma unroll
        for (int u = 0; u < 8; ++u) {
          int k = kb + kk + u;
          pack[u] = (_Float16)fast_tanh(xv * sW1[k] + sB1[k]);
        }
        *(v8h*)(&h1[m * LDSK + kb + kk]) = pack;
      }
    }

    // ---- stage W3[i] -> sWB + b3 (overlaps with layer-2 WMMA across waves) ----
    stageW(W3 + i * H * H, sWB);
    if (tid < H) sB3[tid] = b3[i * H + tid];

    // ---- layer 2: h2 = tanh(h1 @ W2 + b2), WMMA f16 -> f32 (uses sWA) ----
    #pragma unroll
    for (int n0 = 0; n0 < H; n0 += 16) {
      float bias = sB2[n0 + lidx];
      v8f acc;
      #pragma unroll
      for (int r = 0; r < 8; ++r) acc[r] = bias;
      #pragma unroll
      for (int k0 = 0; k0 < H; k0 += 32) {
        FragH a, bm;
        const _Float16* ap = &h1[lidx * LDSK + k0 + 8 * half];   // A: row M=lidx
        a.h[0] = *(const v8h*)(ap);                              // K 0..7 (of half)
        a.h[1] = *(const v8h*)(ap + 16);                         // K 16..23 (of half)
        const _Float16* bp = &sWA[(n0 + lidx) * LDSK + k0 + 8 * half]; // B: col N=lidx
        bm.h[0] = *(const v8h*)(bp);                             // K 0..7 (of half)
        bm.h[1] = *(const v8h*)(bp + 16);                        // K 16..23 (of half)
        acc = __builtin_amdgcn_wmma_f32_16x16x32_f16(
                  false, a.v, false, bm.v, (short)0, acc, false, false);
      }
      #pragma unroll
      for (int r = 0; r < 8; ++r) {
        int m = r + 8 * half;                  // C/D layout: M = r + 8*lane[4]
        h2[m * LDSK + n0 + lidx] = (_Float16)fast_tanh(acc[r]);
      }
    }

    __syncthreads();   // sWB staged & visible; all waves done reading sWA

    // ---- stage W2[i+1] -> sWA + next params (overlaps with layer-3 WMMA) ----
    if (i < 7) {
      stageW(W2 + (i + 1) * H * H, sWA);
      if (tid < H) {
        sW1[tid] = W1[(i + 1) * H + tid];
        sB1[tid] = b1[(i + 1) * H + tid];
        sB2[tid] = b2[(i + 1) * H + tid];
      }
    }

    // ---- layer 3: o = h2 @ W3 + b3; prod *= o (uses sWB) ----
    #pragma unroll
    for (int t = 0; t < 8; ++t) {
      int n0 = t * 16;
      float bias = sB3[n0 + lidx];
      v8f acc;
      #pragma unroll
      for (int r = 0; r < 8; ++r) acc[r] = bias;
      #pragma unroll
      for (int k0 = 0; k0 < H; k0 += 32) {
        FragH a, bm;
        const _Float16* ap = &h2[lidx * LDSK + k0 + 8 * half];
        a.h[0] = *(const v8h*)(ap);
        a.h[1] = *(const v8h*)(ap + 16);
        const _Float16* bp = &sWB[(n0 + lidx) * LDSK + k0 + 8 * half];
        bm.h[0] = *(const v8h*)(bp);
        bm.h[1] = *(const v8h*)(bp + 16);
        acc = __builtin_amdgcn_wmma_f32_16x16x32_f16(
                  false, a.v, false, bm.v, (short)0, acc, false, false);
      }
      #pragma unroll
      for (int r = 0; r < 8; ++r) prod[t][r] *= acc[r];
    }

    __syncthreads();   // sWA staged for i+1; all waves done reading sWB
  }

  // ---- reduce: out[row][j] = sum over rank (n in [32j,32j+32)) of prod ----
  // n = t*16 + lidx, so out j spans tiles 2j,2j+1 and all 16 lidx values.
  #pragma unroll
  for (int r = 0; r < 8; ++r) {
    int m = r + 8 * half;
    #pragma unroll
    for (int j = 0; j < 4; ++j) {
      float s = prod[2 * j][r] + prod[2 * j + 1][r];
      s += __shfl_xor(s, 1);       // xor masks <= 8 stay within each 16-lane half
      s += __shfl_xor(s, 2);
      s += __shfl_xor(s, 4);
      s += __shfl_xor(s, 8);
      if (lidx == 0) out[(size_t)(rowBase + m) * 4 + j] = s;
    }
  }
}

extern "C" void kernel_launch(void* const* d_in, const int* in_sizes, int n_in,
                              void* d_out, int out_size, void* d_ws, size_t ws_size,
                              hipStream_t stream) {
  const float* features = (const float*)d_in[0];
  const float* W1 = (const float*)d_in[1];
  const float* b1 = (const float*)d_in[2];
  const float* W2 = (const float*)d_in[3];
  const float* b2 = (const float*)d_in[4];
  const float* W3 = (const float*)d_in[5];
  const float* b3 = (const float*)d_in[6];
  float* out = (float*)d_out;

  const int B = in_sizes[0] / 8;            // 131072
  const int rowsPerWG = WAVES * 16;         // 128
  dim3 grid(B / rowsPerWG);                 // 1024
  dim3 block(256);
  sepnn_kernel<<<grid, block, 0, stream>>>(features, W1, b1, W2, b2, W3, b3, out);
}
